// HierarchicalMultiHeadAttention_25202868093589
// MI455X (gfx1250) — matrix operand order repaired and checked by reading, not compiled, not run on hardware
//
#include <hip/hip_runtime.h>

#define B_  4
#define C_  64
#define N_  16384
#define H_  8
#define BN_ (B_ * N_)   // 65536

typedef __attribute__((ext_vector_type(16))) _Float16 v16h;
typedef __attribute__((ext_vector_type(8)))  float    v8f;

__device__ __forceinline__ v8f wmma_f16(v16h a, v16h b, v8f c) {
  // D = A(16x32) * B(32x16) + C, f32 accumulate
  return __builtin_amdgcn_wmma_f32_16x16x32_f16(false, a, false, b, (short)0, c, false, false);
}

// -------- gates softmax over 4 levels, per channel --------
__global__ void k_gates(const float* __restrict__ gates, float* __restrict__ gsm) {
  int c = threadIdx.x;
  if (c >= C_) return;
  float v0 = gates[0*C_+c], v1 = gates[1*C_+c], v2 = gates[2*C_+c], v3 = gates[3*C_+c];
  float m  = fmaxf(fmaxf(v0, v1), fmaxf(v2, v3));
  float e0 = __expf(v0-m), e1 = __expf(v1-m), e2 = __expf(v2-m), e3 = __expf(v3-m);
  float s  = e0 + e1 + e2 + e3;
  gsm[0*C_+c] = e0/s; gsm[1*C_+c] = e1/s; gsm[2*C_+c] = e2/s; gsm[3*C_+c] = e3/s;
}

// -------- (B,C,N) f32 -> (B*N, C) f16 rows (both x and orig) --------
__global__ void k_to_nc_f16(const float* __restrict__ x, const float* __restrict__ orig,
                            _Float16* __restrict__ xh, _Float16* __restrict__ oh) {
  size_t tid = (size_t)blockIdx.x * blockDim.x + threadIdx.x;
  if (tid >= (size_t)B_ * C_ * N_) return;
  int n = (int)(tid % N_);
  int c = (int)((tid / N_) % C_);
  int b = (int)(tid / ((size_t)N_ * C_));
  size_t dst = ((size_t)(b * N_ + n)) * C_ + c;
  xh[dst] = (_Float16)x[tid];
  oh[dst] = (_Float16)orig[tid];
}

// -------- W (R x S) f32 row-major -> Wt (S x R) f16 row-major --------
__global__ void k_wt(const float* __restrict__ w, _Float16* __restrict__ wt, int R, int S) {
  int tid = blockIdx.x * blockDim.x + threadIdx.x;
  if (tid >= R * S) return;
  int r = tid / S, s = tid % S;
  wt[(size_t)s * R + r] = (_Float16)w[tid];
}

// -------- input projection: (BN x 64) f16 @ Wt(hd x 64)^T -> f16 --------
// Each wave: 4 M-tiles, B (weights) held in registers across the M loop.
// vt_mode 0: out[m*hd + col]   (row-major, for Q/K)
// vt_mode 1: out[col*BN + m]   (transposed, for V so P*V B-operand is contiguous)
__global__ __launch_bounds__(128) void k_gemm_in(
    const _Float16* __restrict__ X, const _Float16* __restrict__ Wt,
    _Float16* __restrict__ out, int hd, int vt_mode) {
  int lane = threadIdx.x & 31;
  int w    = threadIdx.x >> 5;
  int mb   = blockIdx.x * 256 + w * 64;
  int c0   = blockIdx.y * 16;
  int rw   = lane & 15;
  int kbA  = (lane >> 4) * 8;    // A layout: K = {kbA..kbA+7} U {16+kbA..}
  int kbB  = (lane >> 4) * 8;    // B layout: K = {kbB..kbB+7} U {16+kbB..}
  const _Float16* wr = Wt + (size_t)(c0 + rw) * C_;
  v16h b0, b1;
#pragma unroll
  for (int e = 0; e < 16; ++e) { b0[e] = wr[kbB + e + (e & 8)]; b1[e] = wr[32 + kbB + e + (e & 8)]; }
  int ih  = (lane < 16) ? 0 : 8;
  int col = c0 + (lane & 15);
#pragma unroll
  for (int mt = 0; mt < 4; ++mt) {
    int m0 = mb + mt * 16;
    const _Float16* xr = X + (size_t)(m0 + rw) * C_;
    if (mt < 3) __builtin_prefetch(xr + 16 * C_, 0, 1);
    v16h a0, a1;
#pragma unroll
    for (int j = 0; j < 8; ++j) {
      a0[j]     = xr[kbA + j];
      a0[8 + j] = xr[16 + kbA + j];
      a1[j]     = xr[32 + kbA + j];
      a1[8 + j] = xr[48 + kbA + j];
    }
    v8f acc = {};
    acc = wmma_f16(a0, b0, acc);
    acc = wmma_f16(a1, b1, acc);
    if (!vt_mode) {
#pragma unroll
      for (int r = 0; r < 8; ++r)
        out[(size_t)(m0 + r + ih) * hd + col] = (_Float16)acc[r];
    } else {
#pragma unroll
      for (int r = 0; r < 8; ++r)
        out[(size_t)col * BN_ + m0 + r + ih] = (_Float16)acc[r];
    }
  }
}

// -------- output projection + gated accumulate into f32 (BN x C) --------
// Templated on K-chunk count so all weight operands stay resident in VGPRs.
template <int KC>
__global__ __launch_bounds__(128) void k_gemm_out(
    const _Float16* __restrict__ A, const _Float16* __restrict__ WoT,
    float* __restrict__ Yacc,
    const float* __restrict__ scale, const float* __restrict__ gate) {
  constexpr int HD = KC * 32;
  int lane = threadIdx.x & 31;
  int w    = threadIdx.x >> 5;
  int mb   = blockIdx.x * 256 + w * 64;
  int c0   = blockIdx.y * 16;
  int rw   = lane & 15;
  int kbA  = (lane >> 4) * 8;
  int kbB  = (lane >> 4) * 8;
  const _Float16* wr = WoT + (size_t)(c0 + rw) * HD;
  v16h b[KC];
#pragma unroll
  for (int kc = 0; kc < KC; ++kc)
#pragma unroll
    for (int e = 0; e < 16; ++e) b[kc][e] = wr[kc*32 + kbB + e + (e & 8)];
  int ih  = (lane < 16) ? 0 : 8;
  int col = c0 + (lane & 15);
  float sg = scale[0] * gate[col];
#pragma unroll
  for (int mt = 0; mt < 4; ++mt) {
    int m0 = mb + mt * 16;
    const _Float16* ar = A + (size_t)(m0 + rw) * HD;
    if (mt < 3) __builtin_prefetch(ar + 16 * HD, 0, 1);
    v8f acc = {};
#pragma unroll
    for (int kc = 0; kc < KC; ++kc) {
      v16h a;
#pragma unroll
      for (int j = 0; j < 8; ++j) {
        a[j]     = ar[kc*32 + kbA + j];
        a[8 + j] = ar[kc*32 + 16 + kbA + j];
      }
      acc = wmma_f16(a, b[kc], acc);
    }
#pragma unroll
    for (int r = 0; r < 8; ++r)
      Yacc[(size_t)(m0 + r + ih) * C_ + col] += sg * acc[r];
  }
}

// -------- windowed multi-head attention, one wave per (b, h, 16-row tile) --------
template <int G, int D>
__global__ __launch_bounds__(32) void k_attn(
    const _Float16* qh, const _Float16* __restrict__ kh,
    const _Float16* __restrict__ vt, _Float16* oattn) {
  constexpr int S   = (G > 16) ? (G / 16) : 1;  // key tiles (16 wide)
  constexpr int KT  = S * 16;                   // key span
  constexpr int NC  = (KT + 31) / 32;           // K chunks for P*V
  constexpr int NT2 = (D + 15) / 16;            // output column tiles
  constexpr int HD  = H_ * D;
  __shared__ _Float16 plds[16 * KT];

  int lane  = threadIdx.x;
  int bid   = blockIdx.x;
  int ntile = bid & (N_/16 - 1);
  int h     = (bid / (N_/16)) & (H_ - 1);
  int b     = bid / ((N_/16) * H_);
  int n0    = ntile * 16;
  int nw    = (G >= 16) ? ((n0 / G) * G) : n0;  // key-window start
  int rw    = lane & 15;
  int kbA   = (lane >> 4) * 8;
  int kbB   = (lane >> 4) * 8;
  int ih    = (lane < 16) ? 0 : 8;

  // Q tile in A layout (zero-padded K when D < 32)
  v16h a = {};
  {
    const _Float16* qr = qh + (size_t)(b*N_ + n0 + rw) * HD + h * D;
#pragma unroll
    for (int j = 0; j < 8; ++j) {
      if (kbA + j < D)      a[j]     = qr[kbA + j];
      if (16 + kbA + j < D) a[8 + j] = qr[16 + kbA + j];
    }
  }

  // scores = Q * K^T  (B-operand: lane = key index, K = feature)
  float s[S][8];
#pragma unroll
  for (int t = 0; t < S; ++t) {
    const _Float16* kr = kh + (size_t)(b*N_ + nw + t*16 + rw) * HD + h * D;
    v16h bt = {};
#pragma unroll
    for (int e = 0; e < 16; ++e)
      if (kbB + e + (e & 8) < D) bt[e] = kr[kbB + e + (e & 8)];
    v8f c = {};
    c = wmma_f16(a, bt, c);
#pragma unroll
    for (int r = 0; r < 8; ++r) s[t][r] = c[r];
  }

  // scale + block-diagonal mask (only needed when G < 16)
  const float isc = 1.0f / sqrtf((float)D);
#pragma unroll
  for (int t = 0; t < S; ++t)
#pragma unroll
    for (int r = 0; r < 8; ++r) {
      float v = s[t][r] * isc;
      if (G < 16) {
        int i = r + ih, j = lane & 15;
        if ((i / G) != (j / G)) v = -1e30f;
      }
      s[t][r] = v;
    }

  // row softmax: reduce across 16-lane halves (wave32) with shfl_xor
  float sm[8];
#pragma unroll
  for (int r = 0; r < 8; ++r) {
    float m = s[0][r];
#pragma unroll
    for (int t = 1; t < S; ++t) m = fmaxf(m, s[t][r]);
#pragma unroll
    for (int o = 8; o >= 1; o >>= 1) m = fmaxf(m, __shfl_xor(m, o, 32));
    float acc = 0.0f;
#pragma unroll
    for (int t = 0; t < S; ++t) { float p = __expf(s[t][r] - m); s[t][r] = p; acc += p; }
#pragma unroll
    for (int o = 8; o >= 1; o >>= 1) acc += __shfl_xor(acc, o, 32);
    sm[r] = acc;
  }

  // bounce P through LDS to convert C-layout -> A-layout
#pragma unroll
  for (int t = 0; t < S; ++t)
#pragma unroll
    for (int r = 0; r < 8; ++r)
      plds[(r + ih) * KT + t*16 + (lane & 15)] = (_Float16)s[t][r];
  __syncthreads();

  v16h ap[NC];
#pragma unroll
  for (int kc = 0; kc < NC; ++kc) {
    v16h t = {};
#pragma unroll
    for (int j = 0; j < 8; ++j) {
      int k0 = kc*32 + kbA + j;
      int k1 = kc*32 + 16 + kbA + j;
      if (k0 < KT) t[j]     = plds[rw * KT + k0];
      if (k1 < KT) t[8 + j] = plds[rw * KT + k1];
    }
    ap[kc] = t;
  }

  // O = P * V ; V stored transposed [hd][BN] so per-lane loads are contiguous
#pragma unroll
  for (int nt = 0; nt < NT2; ++nt) {
    int ddg = nt*16 + (lane & 15);
    const _Float16* vr = vt + ((size_t)(h*D + (ddg < D ? ddg : 0))) * BN_ + b*N_ + nw;
    v8f oc = {};
#pragma unroll
    for (int kc = 0; kc < NC; ++kc) {
      v16h bv = {};
#pragma unroll
      for (int e = 0; e < 16; ++e) {
        int k = kc*32 + kbB + e + (e & 8);
        if (ddg < D && k < KT) bv[e] = vr[k];
      }
      oc = wmma_f16(ap[kc], bv, oc);
    }
    if (ddg < D) {
#pragma unroll
      for (int r = 0; r < 8; ++r)
        oattn[(size_t)(b*N_ + n0 + r + ih) * HD + h*D + ddg] = (_Float16)(oc[r] / sm[r]);
    }
  }
}

// -------- (BN x C) f32 -> (B, C, N) f32 output --------
__global__ void k_final(const float* __restrict__ Yacc, float* __restrict__ out) {
  size_t tid = (size_t)blockIdx.x * blockDim.x + threadIdx.x;
  if (tid >= (size_t)B_ * C_ * N_) return;
  int n = (int)(tid % N_);
  int c = (int)((tid / N_) % C_);
  int b = (int)(tid / ((size_t)N_ * C_));
  out[tid] = Yacc[((size_t)(b * N_ + n)) * C_ + c];
}

extern "C" void kernel_launch(void* const* d_in, const int* in_sizes, int n_in,
                              void* d_out, int out_size, void* d_ws, size_t ws_size,
                              hipStream_t stream) {
  (void)in_sizes; (void)n_in; (void)out_size; (void)ws_size;
  const float* x     = (const float*)d_in[0];
  const float* orig  = (const float*)d_in[1];
  const float* gates = (const float*)d_in[2];

  char* ws = (char*)d_ws;
  size_t off = 0;
  auto alloc = [&](size_t bytes) -> char* {
    char* p = ws + off; off += (bytes + 255) & ~(size_t)255; return p;
  };
  float*    gsm  = (float*)   alloc((size_t)4 * C_ * sizeof(float));
  _Float16* xh   = (_Float16*)alloc((size_t)BN_ * C_  * 2);
  _Float16* qh   = (_Float16*)alloc((size_t)BN_ * 256 * 2);   // Q, reused as attention output
  _Float16* oh   = (_Float16*)alloc((size_t)BN_ * C_  * 2);
  _Float16* kh   = (_Float16*)alloc((size_t)BN_ * 256 * 2);
  _Float16* vt   = (_Float16*)alloc((size_t)BN_ * 256 * 2);
  float*    Yacc = (float*)   alloc((size_t)BN_ * C_  * 4);
  _Float16* wqt  = (_Float16*)alloc((size_t)256 * 64 * 2);
  _Float16* wkt  = (_Float16*)alloc((size_t)256 * 64 * 2);
  _Float16* wvt  = (_Float16*)alloc((size_t)256 * 64 * 2);
  _Float16* wot  = (_Float16*)alloc((size_t)256 * 64 * 2);

  k_gates<<<1, 64, 0, stream>>>(gates, gsm);
  {
    int total = B_ * C_ * N_;
    k_to_nc_f16<<<(total + 255) / 256, 256, 0, stream>>>(x, orig, xh, oh);
  }
  hipMemsetAsync(Yacc, 0, (size_t)BN_ * C_ * sizeof(float), stream);

  static const int UG[8] = {128, 32, 64, 16, 32, 8, 16, 4};
  static const int UD[8] = {4, 4, 16, 16, 16, 32, 16, 32};
  static const int UL[8] = {0, 0, 1, 1, 2, 2, 3, 3};

  for (int u = 0; u < 8; ++u) {
    int d = UD[u], lvl = UL[u];
    int hd = H_ * d;
    // pytree flatten order per unit (sorted dict keys): Wk, Wo, Wq, Wv, scale
    int base = 3 + u * 5;
    const float* Wk    = (const float*)d_in[base + 0];
    const float* Wo    = (const float*)d_in[base + 1];
    const float* Wq    = (const float*)d_in[base + 2];
    const float* Wv    = (const float*)d_in[base + 3];
    const float* scale = (const float*)d_in[base + 4];

    int wtn = C_ * hd;
    k_wt<<<(wtn + 255) / 256, 256, 0, stream>>>(Wq, wqt, C_, hd);
    k_wt<<<(wtn + 255) / 256, 256, 0, stream>>>(Wk, wkt, C_, hd);
    k_wt<<<(wtn + 255) / 256, 256, 0, stream>>>(Wv, wvt, C_, hd);
    k_wt<<<(wtn + 255) / 256, 256, 0, stream>>>(Wo, wot, hd, C_);

    dim3 gi(BN_ / 256, hd / 16);
    k_gemm_in<<<gi, 128, 0, stream>>>(xh, wqt, qh, hd, 0);
    k_gemm_in<<<gi, 128, 0, stream>>>(oh, wkt, kh, hd, 0);
    k_gemm_in<<<gi, 128, 0, stream>>>(oh, wvt, vt, hd, 1);

    int ablocks = B_ * H_ * (N_ / 16);
    switch (u) {
      case 0: k_attn<128, 4><<<ablocks, 32, 0, stream>>>(qh, kh, vt, qh); break;
      case 1: k_attn< 32, 4><<<ablocks, 32, 0, stream>>>(qh, kh, vt, qh); break;
      case 2: k_attn< 64,16><<<ablocks, 32, 0, stream>>>(qh, kh, vt, qh); break;
      case 3: k_attn< 16,16><<<ablocks, 32, 0, stream>>>(qh, kh, vt, qh); break;
      case 4: k_attn< 32,16><<<ablocks, 32, 0, stream>>>(qh, kh, vt, qh); break;
      case 5: k_attn<  8,32><<<ablocks, 32, 0, stream>>>(qh, kh, vt, qh); break;
      case 6: k_attn< 16,16><<<ablocks, 32, 0, stream>>>(qh, kh, vt, qh); break;
      case 7: k_attn<  4,32><<<ablocks, 32, 0, stream>>>(qh, kh, vt, qh); break;
    }

    dim3 go(BN_ / 256, C_ / 16);
    switch (hd / 32) {
      case 1: k_gemm_out<1><<<go, 128, 0, stream>>>(qh, wot, Yacc, scale, gsm + lvl * C_); break;
      case 4: k_gemm_out<4><<<go, 128, 0, stream>>>(qh, wot, Yacc, scale, gsm + lvl * C_); break;
      case 8: k_gemm_out<8><<<go, 128, 0, stream>>>(qh, wot, Yacc, scale, gsm + lvl * C_); break;
    }
  }

  {
    int total = B_ * C_ * N_;
    k_final<<<(total + 255) / 256, 256, 0, stream>>>(Yacc, (float*)d_out);
  }
}
